// Rudder_Net_3736621548129
// MI455X (gfx1250) — matrix operand order repaired and checked by reading, not compiled, not run on hardware
//
#include <hip/hip_runtime.h>
#include <hip/hip_bf16.h>

typedef __attribute__((ext_vector_type(16))) _Float16 v16h;
typedef __attribute__((ext_vector_type(8)))  float    v8f;

#define B_DIM      64
#define T_LEN      512
#define BT_TOTAL   (B_DIM * T_LEN)
#define TILE       32          // samples per workgroup in encoder
#define K1_THREADS 64          // 2 waves

// ---------------- shared memory layout (bytes), encoder kernel -------------
// total 163072 B  (< 160 KiB conservative cap; ISA allows up to 320 KB/WG)
#define SM_WCONV   0           // staged f32 conv/misc weights: 6382 floats
#define SM_FCW16   25600       // packed B-fragments: obs_fc 16 tiles + w_ci 12 tiles, f16
#define SM_C1      54272       // conv1 out, 16 samples * 675 f32 = 43200 B; reused as conv3 out (32*256*4=32768)
#define SM_C2      97536       // conv2 out, 32*512*4 = 65536 B; reused as feats (16384) + enc f16 (6144)
#define SMEM_BYTES 163072

// float offsets inside SM_WCONV region
#define OFF_W1 0       // 27
#define OFF_B1 27      // 3
#define OFF_W2 30      // 216
#define OFF_B2 246     // 8
#define OFF_W3 254     // 1152
#define OFF_B3 1406    // 16
#define OFF_W4 1422    // 4608
#define OFF_B4 6030    // 32
#define OFF_AW 6062    // act_fc_w 160
#define OFF_AB 6222    // act_fc_b 32
#define OFF_FB 6254    // obs_fc_b 64
#define OFF_CB 6318    // b_ci 64      -> end 6382 floats = 25528 B

// ---------------------------------------------------------------------------
// WMMA helpers (v_wmma_f32_16x16x32_f16, wave32, D = A(16x32) * B(32x16) + C)
// ---------------------------------------------------------------------------
__device__ inline v8f wmma_f16(v16h a, v16h b, v8f c) {
  return __builtin_amdgcn_wmma_f32_16x16x32_f16(
      /*neg_a=*/false, a, /*neg_b=*/false, b,
      /*c_mod=*/(short)0, c, /*reuse_a=*/false, /*reuse_b=*/false);
}

// A-matrix 16x32 f16 layout: lanes 0-15 row M=lane, K0..15;
// lanes 16-31 same rows, K16..31 (contiguous, matches packed B).
// Loads are per-lane contiguous (8+8 elements) -> ds_load_b128 pairs.
__device__ inline v16h load_A_f32(const float* __restrict__ src, int ld, int k0, int lane) {
  const int row = lane & 15;
  const int kb  = (lane >> 4) << 4;             // 0 or 16
  const float* p = src + row * ld + k0 + kb;
  v16h a;
#pragma unroll
  for (int i = 0; i < 8; ++i) a[i] = (_Float16)p[i];
#pragma unroll
  for (int i = 0; i < 8; ++i) a[8 + i] = (_Float16)p[8 + i];
  return a;
}

__device__ inline v16h load_A_h(const _Float16* __restrict__ src, int ld, int k0, int lane) {
  const int row = lane & 15;
  const int kb  = (lane >> 4) << 4;
  const _Float16* p = src + row * ld + k0 + kb;
  v16h a;
#pragma unroll
  for (int i = 0; i < 8; ++i) a[i] = p[i];
#pragma unroll
  for (int i = 0; i < 8; ++i) a[8 + i] = p[8 + i];
  return a;
}

// B-matrix 32x16 f16 fragment gather from row-major f32 W[K][N] (used only for
// one-time register staging in the scan kernel).
__device__ inline v16h load_B_f32(const float* __restrict__ w, int ld, int k0, int n0, int lane) {
  const int n  = n0 + (lane & 15);
  const int kb = k0 + ((lane >> 4) << 4);       // 0 or 16
  v16h b;
#pragma unroll
  for (int i = 0; i < 16; ++i) b[i] = (_Float16)w[(kb + i) * ld + n];
  return b;
}

// Packed-B fragment: 32 contiguous bytes per lane -> 2x ds_load_b128.
__device__ inline v16h load_B_packed(const _Float16* __restrict__ packed, int tile, int lane) {
  return *(const v16h*)(packed + (((tile << 5) + lane) << 4));
}

// ---------------------------------------------------------------------------
// Async global->LDS staging (gfx1250 GLOBAL_LOAD_ASYNC_TO_LDS_B32, ASYNCcnt)
// ---------------------------------------------------------------------------
__device__ inline void async_cp_f32(const float* __restrict__ g, float* l, int n,
                                    int tid, int nthr) {
  for (int i = tid; i < n; i += nthr) {
    unsigned ldsAddr = (unsigned)(unsigned long long)(l + i); // low 32b of LDS aperture = byte offset
    const float* gp = g + i;
    asm volatile("global_load_async_to_lds_b32 %0, %1, off"
                 :: "v"(ldsAddr), "v"(gp) : "memory");
  }
}

__device__ inline float sigmoidf_fast(float x) {
  return 1.f / (1.f + __expf(-x));
}

// ---------------------------------------------------------------------------
// Cooperative 3x3 stride-2 pad-1 conv + ReLU over S samples (NCHW per sample)
// ---------------------------------------------------------------------------
__device__ inline void conv3x3_s2_relu(
    const float* __restrict__ in, int inStride, int Ci, int Hi, int Wi,
    float* __restrict__ out, int outStride, int Co, int Ho, int Wo,
    const float* __restrict__ w, const float* __restrict__ b,
    int S, int tid, int nthr)
{
  const int total = S * Co * Ho * Wo;
  for (int idx = tid; idx < total; idx += nthr) {
    int ow  = idx % Wo;
    int tmp = idx / Wo;
    int oh  = tmp % Ho; tmp /= Ho;
    int co  = tmp % Co;
    int s   = tmp / Co;
    float acc = b[co];
    const float* ip = in + s * inStride;
    const float* wp = w + co * Ci * 9;
    for (int ci = 0; ci < Ci; ++ci) {
      const float* ic = ip + ci * Hi * Wi;
      const float* wc = wp + ci * 9;
#pragma unroll
      for (int kh = 0; kh < 3; ++kh) {
        int ih = oh * 2 - 1 + kh;
        if (ih < 0 || ih >= Hi) continue;
#pragma unroll
        for (int kw = 0; kw < 3; ++kw) {
          int iw = ow * 2 - 1 + kw;
          if (iw < 0 || iw >= Wi) continue;
          acc += ic[ih * Wi + iw] * wc[kh * 3 + kw];
        }
      }
    }
    out[s * outStride + co * Ho * Wo + oh * Wo + ow] = fmaxf(acc, 0.f);
  }
}

// ===========================================================================
// Kernel 1: conv encoder + FC encoders + ci = tanh(enc @ w_ci + b_ci)
// One 64-thread block (2 waves) per 32-sample tile; each wave owns M=16 rows
// for the WMMA stages.
// ===========================================================================
__global__ __launch_bounds__(K1_THREADS)
void rudder_encoder(const float* __restrict__ obs,  const float* __restrict__ acts,
                    const float* __restrict__ c1w,  const float* __restrict__ c1b,
                    const float* __restrict__ c2w,  const float* __restrict__ c2b,
                    const float* __restrict__ c3w,  const float* __restrict__ c3b,
                    const float* __restrict__ c4w,  const float* __restrict__ c4b,
                    const float* __restrict__ ofw,  const float* __restrict__ ofb,
                    const float* __restrict__ afw,  const float* __restrict__ afb,
                    const float* __restrict__ wci,  const float* __restrict__ bci,
                    float* __restrict__ ci_out)
{
  __shared__ __align__(32) char smem[SMEM_BYTES];
  const int tid  = threadIdx.x;
  const int lane = tid & 31;
  const int wave = tid >> 5;
  const int lo   = lane & 15;
  const int hi   = lane >> 4;
  const int tileBase = blockIdx.x * TILE;

  float*     wconv = (float*)(smem + SM_WCONV);
  _Float16*  fcwP  = (_Float16*)(smem + SM_FCW16);   // obs_fc: 16 packed B tiles
  _Float16*  ciwP  = fcwP + 16 * 32 * 16;            // w_ci : 12 packed B tiles
  float*     c1    = (float*)(smem + SM_C1);
  float*     c2    = (float*)(smem + SM_C2);

  // L2 prefetch of this tile's observations (global_prefetch_b8)
  {
    const char* obsTile = (const char*)(obs + (size_t)tileBase * 900);
    for (int off = tid * 256; off < TILE * 900 * 4; off += K1_THREADS * 256)
      __builtin_prefetch(obsTile + off, 0, 1);
  }

  // ---- stage conv + small-FC weights into LDS asynchronously -------------
  async_cp_f32(c1w, wconv + OFF_W1,   27, tid, K1_THREADS);
  async_cp_f32(c1b, wconv + OFF_B1,    3, tid, K1_THREADS);
  async_cp_f32(c2w, wconv + OFF_W2,  216, tid, K1_THREADS);
  async_cp_f32(c2b, wconv + OFF_B2,    8, tid, K1_THREADS);
  async_cp_f32(c3w, wconv + OFF_W3, 1152, tid, K1_THREADS);
  async_cp_f32(c3b, wconv + OFF_B3,   16, tid, K1_THREADS);
  async_cp_f32(c4w, wconv + OFF_W4, 4608, tid, K1_THREADS);
  async_cp_f32(c4b, wconv + OFF_B4,   32, tid, K1_THREADS);
  async_cp_f32(afw, wconv + OFF_AW,  160, tid, K1_THREADS);
  async_cp_f32(afb, wconv + OFF_AB,   32, tid, K1_THREADS);
  async_cp_f32(ofb, wconv + OFF_FB,   64, tid, K1_THREADS);
  async_cp_f32(bci, wconv + OFF_CB,   64, tid, K1_THREADS);

  // ---- pack FC weights into WMMA B-fragment order (f16) ------------------
  // Tile t of matrix W[K][64]: n = (t&3)*16 + (lane&15), k = (t>>2)*32 + 16*(lane>>4) + i.
  // Each (tile,lane) fragment is 16 contiguous halfs => loads back as 2x ds_load_b128.
  // Lane-adjacent n => global reads coalesce.
  for (int idx = tid; idx < (16 + 12) * 32; idx += K1_THREADS) {
    const int t    = idx >> 5;
    const int ln   = idx & 31;
    const bool isC = (t >= 16);
    const int tt   = isC ? (t - 16) : t;
    const float* src = isC ? wci : ofw;
    _Float16*   dst  = (isC ? ciwP : fcwP) + (((tt << 5) + ln) << 4);
    const int n  = (tt & 3) * 16 + (ln & 15);
    const int kb = (tt >> 2) * 32 + ((ln >> 4) << 4);
#pragma unroll
    for (int i = 0; i < 16; ++i) dst[i] = (_Float16)src[(kb + i) * 64 + n];
  }

  asm volatile("s_wait_asynccnt 0x0" ::: "memory");
  __syncthreads();

  // ---- conv stack (VALU f32; channels too small for WMMA) ----------------
  // conv1+conv2 in two 16-sample halves so the c1 buffer stays small
  for (int h = 0; h < 2; ++h) {
    const float* inObs = obs + (size_t)(tileBase + h * 16) * 900;
    conv3x3_s2_relu(inObs, 900, 1, 30, 30, c1, 675, 3, 15, 15,
                    wconv + OFF_W1, wconv + OFF_B1, 16, tid, K1_THREADS);
    __syncthreads();
    conv3x3_s2_relu(c1, 675, 3, 15, 15, c2 + h * 16 * 512, 512, 8, 8, 8,
                    wconv + OFF_W2, wconv + OFF_B2, 16, tid, K1_THREADS);
    __syncthreads();
  }
  float* c3 = (float*)(smem + SM_C1);              // overlay (c1 dead)
  conv3x3_s2_relu(c2, 512, 8, 8, 8, c3, 256, 16, 4, 4,
                  wconv + OFF_W3, wconv + OFF_B3, TILE, tid, K1_THREADS);
  __syncthreads();
  float* feats = (float*)(smem + SM_C2);           // overlay (c2 dead), [32][128]
  conv3x3_s2_relu(c3, 256, 16, 4, 4, feats, 128, 32, 2, 2,
                  wconv + OFF_W4, wconv + OFF_B4, TILE, tid, K1_THREADS);
  __syncthreads();

  // ---- obs_fc: (16x128) @ (128x64) per wave via WMMA ---------------------
  _Float16* enc = (_Float16*)(smem + SM_C2 + TILE * 128 * 4);   // [32][96] f16
  const float* myfeats = feats + wave * 16 * 128;

  {
    v16h A[4];
#pragma unroll
    for (int kt = 0; kt < 4; ++kt) A[kt] = load_A_f32(myfeats, 128, kt * 32, lane);
#pragma unroll
    for (int nt = 0; nt < 4; ++nt) {
      float bv = wconv[OFF_FB + nt * 16 + lo];
      v8f acc;
#pragma unroll
      for (int r = 0; r < 8; ++r) acc[r] = bv;
#pragma unroll
      for (int kt = 0; kt < 4; ++kt)
        acc = wmma_f16(A[kt], load_B_packed(fcwP, kt * 4 + nt, lane), acc);
#pragma unroll
      for (int r = 0; r < 8; ++r) {
        int m = r + (hi << 3);
        enc[(wave * 16 + m) * 96 + nt * 16 + lo] = (_Float16)fmaxf(acc[r], 0.f);
      }
    }
  }

  // ---- act_fc: K=5 is tiny -> scalar VALU, per-wave samples --------------
  for (int idx = lane; idx < 16 * 32; idx += 32) {
    int sl = idx >> 5, j = idx & 31;
    int s  = wave * 16 + sl;
    const float* ap = acts + (size_t)(tileBase + s) * 5;
    float sum = wconv[OFF_AB + j];
#pragma unroll
    for (int k = 0; k < 5; ++k) sum += ap[k] * wconv[OFF_AW + k * 32 + j];
    enc[s * 96 + 64 + j] = (_Float16)fmaxf(sum, 0.f);
  }
  __builtin_amdgcn_wave_barrier();
  asm volatile("s_wait_dscnt 0x0" ::: "memory");   // per-wave LDS RAW fence

  // ---- ci = tanh(enc(16x96) @ w_ci(96x64) + b_ci) via WMMA ---------------
  {
    const _Float16* myenc = enc + wave * 16 * 96;
    v16h A[3];
#pragma unroll
    for (int kt = 0; kt < 3; ++kt) A[kt] = load_A_h(myenc, 96, kt * 32, lane);
#pragma unroll
    for (int nt = 0; nt < 4; ++nt) {
      float bv = wconv[OFF_CB + nt * 16 + lo];
      v8f acc;
#pragma unroll
      for (int r = 0; r < 8; ++r) acc[r] = bv;
#pragma unroll
      for (int kt = 0; kt < 3; ++kt)
        acc = wmma_f16(A[kt], load_B_packed(ciwP, kt * 4 + nt, lane), acc);
#pragma unroll
      for (int r = 0; r < 8; ++r) {
        int m   = r + (hi << 3);
        int btI = tileBase + wave * 16 + m;
        ci_out[(size_t)btI * 64 + nt * 16 + lo] = tanhf(acc[r]);
      }
    }
  }
}

// ===========================================================================
// Kernel 2: sequential scan over T.  64 independent batch rows -> 4 waves,
// each wave integrates M=16 rows with the f32 state in WMMA C/D layout.
//   ig = sigmoid(c @ w_ig + b_ig);  c += ci_t * ig;  reward = c @ out_w + b
// ===========================================================================
__global__ __launch_bounds__(128)
void rudder_scan(const float* __restrict__ ci,   const float* __restrict__ wig,
                 const float* __restrict__ big,  const float* __restrict__ outw,
                 const float* __restrict__ outb, float* __restrict__ out)
{
  __shared__ float ctmp[4][16][72];   // per-wave transpose staging (+pad)
  const int tid  = threadIdx.x;
  const int lane = tid & 31;
  const int wave = tid >> 5;
  const int lo   = lane & 15;
  const int hi   = lane >> 4;
  const int rowBase = wave * 16;

  // w_ig B-fragments resident in registers (2 K-tiles x 4 N-tiles, one-time gather)
  v16h Bw[2][4];
#pragma unroll
  for (int kt = 0; kt < 2; ++kt)
#pragma unroll
    for (int nt = 0; nt < 4; ++nt)
      Bw[kt][nt] = load_B_f32(wig, 64, kt * 32, nt * 16, lane);

  float bias_n[4], ow[4];
#pragma unroll
  for (int nt = 0; nt < 4; ++nt) {
    bias_n[nt] = big[nt * 16 + lo];
    ow[nt]     = outw[nt * 16 + lo];   // out_w shape (64,1)
  }
  const float ob = outb[0];

  v8f c[4];
#pragma unroll
  for (int nt = 0; nt < 4; ++nt)
#pragma unroll
    for (int r = 0; r < 8; ++r) c[nt][r] = 0.f;

  for (int t = 0; t < T_LEN; ++t) {
    // prefetch next step's ci slice into L2/WGP$ while this step computes
    {
      const int prow = rowBase + lo;                 // 16 rows per wave
      const char* p  = (const char*)(ci + ((size_t)prow * T_LEN + (t + 1)) * 64);
      __builtin_prefetch(p + hi * 128,      0, 1);   // covers 256B = 64 floats/row
      __builtin_prefetch(p + hi * 128 + 64, 0, 1);
    }

    // D-layout -> A-layout transpose through per-wave LDS slice
#pragma unroll
    for (int nt = 0; nt < 4; ++nt)
#pragma unroll
      for (int r = 0; r < 8; ++r)
        ctmp[wave][r + (hi << 3)][nt * 16 + lo] = c[nt][r];
    __builtin_amdgcn_wave_barrier();
    asm volatile("s_wait_dscnt 0x0" ::: "memory");

    v16h A0 = load_A_f32(&ctmp[wave][0][0], 72, 0,  lane);
    v16h A1 = load_A_f32(&ctmp[wave][0][0], 72, 32, lane);
    __builtin_amdgcn_wave_barrier();

    float p[8];
#pragma unroll
    for (int r = 0; r < 8; ++r) p[r] = 0.f;

#pragma unroll
    for (int nt = 0; nt < 4; ++nt) {
      v8f g;
#pragma unroll
      for (int r = 0; r < 8; ++r) g[r] = bias_n[nt];
      g = wmma_f16(A0, Bw[0][nt], g);
      g = wmma_f16(A1, Bw[1][nt], g);
#pragma unroll
      for (int r = 0; r < 8; ++r) {
        float igv = sigmoidf_fast(g[r]);
        int   row = rowBase + r + (hi << 3);
        float civ = ci[((size_t)row * T_LEN + t) * 64 + nt * 16 + lo];
        float cn  = c[nt][r] + civ * igv;
        c[nt][r]  = cn;
        p[r]     += cn * ow[nt];
      }
    }

    // 64-feature reduction: sum over the 16 lanes of each half-wave
#pragma unroll
    for (int r = 0; r < 8; ++r) {
      float s = p[r];
      s += __shfl_xor(s, 1, 32);
      s += __shfl_xor(s, 2, 32);
      s += __shfl_xor(s, 4, 32);
      s += __shfl_xor(s, 8, 32);
      if (lo == 0) {
        int row = rowBase + r + (hi << 3);
        out[(size_t)row * T_LEN + t] = s + ob;
      }
    }
  }
}

// ===========================================================================
extern "C" void kernel_launch(void* const* d_in, const int* in_sizes, int n_in,
                              void* d_out, int out_size, void* d_ws, size_t ws_size,
                              hipStream_t stream)
{
  const float* obs  = (const float*)d_in[0];
  const float* acts = (const float*)d_in[1];
  const float* c1w  = (const float*)d_in[2];
  const float* c1b  = (const float*)d_in[3];
  const float* c2w  = (const float*)d_in[4];
  const float* c2b  = (const float*)d_in[5];
  const float* c3w  = (const float*)d_in[6];
  const float* c3b  = (const float*)d_in[7];
  const float* c4w  = (const float*)d_in[8];
  const float* c4b  = (const float*)d_in[9];
  const float* ofw  = (const float*)d_in[10];
  const float* ofb  = (const float*)d_in[11];
  const float* afw  = (const float*)d_in[12];
  const float* afb  = (const float*)d_in[13];
  const float* wci  = (const float*)d_in[14];
  const float* bci  = (const float*)d_in[15];
  const float* wig  = (const float*)d_in[16];
  const float* big  = (const float*)d_in[17];
  const float* outw = (const float*)d_in[18];
  const float* outb = (const float*)d_in[19];

  float* ci_ws = (float*)d_ws;   // (B*T, 64) f32 = 8 MB scratch

  rudder_encoder<<<BT_TOTAL / TILE, K1_THREADS, 0, stream>>>(
      obs, acts, c1w, c1b, c2w, c2b, c3w, c3b, c4w, c4b,
      ofw, ofb, afw, afb, wci, bci, ci_ws);

  rudder_scan<<<1, 128, 0, stream>>>(ci_ws, wig, big, outw, outb, (float*)d_out);

  (void)in_sizes; (void)n_in; (void)out_size; (void)ws_size;
}
